// AutoregressiveLSTM_21423296872546
// MI455X (gfx1250) — matrix operand order repaired and checked by reading, not compiled, not run on hardware
//
#include <hip/hip_runtime.h>

typedef __attribute__((ext_vector_type(16))) _Float16 v16h;
typedef __attribute__((ext_vector_type(8)))  _Float16 v8h;
typedef __attribute__((ext_vector_type(8)))  float    v8f;

// ---------------- problem constants ----------------
constexpr int kB = 8192, kS = 168, kF = 96;
constexpr int kT = 24;
constexpr int WAVES = 4;          // waves per workgroup
constexpr int ROWS  = 16;         // batch rows per wave (one WMMA M-tile)
constexpr int NTHR  = WAVES * 32; // wave32

// ---------------- LDS layout (units: _Float16 elements) ----------------
// Weight blocks: each 32(K)x16(N) block stored as 32 lanes x 16 halves (WMMA B-frag layout)
constexpr int OFF_W1  = 0;                   // 96x256  -> 3kt x 16nt = 48 blocks
constexpr int OFF_UH1 = OFF_W1  + 48 * 512;  // 64x256  -> 32 blocks
constexpr int OFF_W2  = OFF_UH1 + 32 * 512;  // 64x128  -> 16 blocks
constexpr int OFF_UH2 = OFF_W2  + 16 * 512;  // 32x128  -> 8 blocks
constexpr int OFF_UHD = OFF_UH2 +  8 * 512;  // 32x128  -> 8 blocks
constexpr int OFF_P1  = OFF_UHD +  8 * 512;  // 32x64   -> 4 blocks
constexpr int OFF_P2  = OFF_P1  +  4 * 512;  // 64x32   -> 4 blocks
constexpr int W_TOT   = OFF_P2  +  4 * 512;  // 61440 halves = 120 KB

// small f32 constants in LDS
constexpr int C_B1 = 0, C_B2 = 256, C_BD = 384, C_PB1 = 512, C_PB2 = 576,
              C_WD = 608, C_WO = 736, C_BO = 768, C_TOT = 772;

// per-wave stage region (3584 halves = 7 KB). Temporal aliasing:
//   encoder phase : [0,512) h2 f16 | [512,1536) h1 f16 | [1536,3584) c1 f32 (lane-packed)
//   decode  phase : [0,512) h2 f16 | [512,1536) x1 f16 | [1536,..) x2 f32 (512) + pred f32 (16)
constexpr int STG = 3584;
constexpr int SH2 = 0;     // 16x32 f16, stride 32
constexpr int SH1 = 512;   // 16x64 f16, stride 64 (decode: x1 stage)
constexpr int SC1 = 1536;  // encoder: c1 = 4 chunks x 32 lanes x v8f ; decode: x2f + pred

// ---------------- helpers ----------------
static __device__ __forceinline__ v8f zero8() {
  v8f z;
#pragma unroll
  for (int i = 0; i < 8; ++i) z[i] = 0.0f;
  return z;
}

static __device__ __forceinline__ v8f wmma32(v16h a, v16h b, v8f c) {
  // D = A(16x32 f16) * B(32x16 f16) + C(16x16 f32)
  return __builtin_amdgcn_wmma_f32_16x16x32_f16(false, a, false, b, (short)0, c,
                                                false, false);
}

// B-fragment: block (kt,nt) of a K x N matrix pre-swizzled in LDS; 32B per lane.
static __device__ __forceinline__ v16h loadB(const _Float16* base, int ktn, int kt,
                                             int nt, int lane) {
  return *(const v16h*)(base + ((nt * ktn + kt) << 9) + (lane << 4));
}

// A-fragment (16x32 f16) from a row-major f16 stage [16][stride].
// lane<16: row=lane, K in {kt*32+0..7, +16..23}; lane>=16: row=lane-16, K {+8..15, +24..31}
static __device__ __forceinline__ v16h load_a(const _Float16* stage, int stride,
                                              int kt, int lane) {
  const _Float16* p = stage + (lane & 15) * stride + (kt << 5) + ((lane & 16) ? 8 : 0);
  v8h lo = *(const v8h*)p;
  v8h hi = *(const v8h*)(p + 16);
  v16h a;
#pragma unroll
  for (int i = 0; i < 8; ++i) { a[i] = lo[i]; a[i + 8] = hi[i]; }
  return a;
}

// A-fragment of the x tile loaded DIRECTLY from global memory (f32 -> f16).
// p0 = start of this lane's batch row at time t. Lane pair (L, L+16) consumes
// the full 384B row in aligned 32B chunks, exactly once.
static __device__ __forceinline__ v16h load_xfrag(const float* __restrict__ p0,
                                                  int kt, int lane) {
  const float* p = p0 + (kt << 5) + ((lane & 16) ? 8 : 0);
  float4 a0 = ((const float4*)p)[0];
  float4 a1 = ((const float4*)p)[1];
  float4 b0 = ((const float4*)(p + 16))[0];
  float4 b1 = ((const float4*)(p + 16))[1];
  v16h r;
  r[0] = (_Float16)a0.x;  r[1] = (_Float16)a0.y;
  r[2] = (_Float16)a0.z;  r[3] = (_Float16)a0.w;
  r[4] = (_Float16)a1.x;  r[5] = (_Float16)a1.y;
  r[6] = (_Float16)a1.z;  r[7] = (_Float16)a1.w;
  r[8] = (_Float16)b0.x;  r[9] = (_Float16)b0.y;
  r[10] = (_Float16)b0.z; r[11] = (_Float16)b0.w;
  r[12] = (_Float16)b1.x; r[13] = (_Float16)b1.y;
  r[14] = (_Float16)b1.z; r[15] = (_Float16)b1.w;
  return r;
}

static __device__ __forceinline__ float sigf(float x) {
  return 1.0f / (1.0f + __expf(-x));
}
static __device__ __forceinline__ float tanhc(float x) {
  x = fminf(fmaxf(x, -15.0f), 15.0f);
  float e = __expf(2.0f * x);
  return (e - 1.0f) / (e + 1.0f);
}

// Swizzle a K x N row-major f32 weight matrix into WMMA B-frag blocks (f16) in LDS.
static __device__ void fill_blocks(_Float16* dst, const float* __restrict__ W,
                                   int K, int N, int tid, int nthr) {
  const int ktn = K >> 5;
  const int total = (ktn * (N >> 4)) << 9;
  for (int i = tid; i < total; i += nthr) {
    const int blk = i >> 9, r = i & 511;
    const int ln = r >> 4, t = r & 15;
    const int nt = blk / ktn, kt = blk - nt * ktn;
    const int k = kt * 32 + ((ln & 16) ? 8 : 0) + (t < 8 ? t : t + 8);   // table order: K {8h+0..7, 16+8h+0..7}
    const int n = nt * 16 + (ln & 15);
    dst[blk * 512 + r] = (_Float16)W[(size_t)k * N + n];
  }
}

static __device__ void copyc(float* dst, const float* __restrict__ src, int n,
                             int tid, int nthr) {
  for (int i = tid; i < n; i += nthr) dst[i] = src[i];
}

// projection head: h2(stage) -> relu(P1) -> relu(P2) -> Wo dot -> pred + global out
static __device__ __forceinline__ void project_store(_Float16* stg,
                                                     const _Float16* wlds,
                                                     const float* clds, int lane,
                                                     float* __restrict__ out,
                                                     int rowBase, int t) {
  const int ncol = lane & 15;
  const int mrow = (lane >> 4) << 3;
  v16h ah2 = load_a(stg + SH2, 32, 0, lane);
#pragma unroll
  for (int nt = 0; nt < 4; ++nt) {
    v8f acc = wmma32(ah2, loadB(wlds + OFF_P1, 1, 0, nt, lane), zero8());
    float pb = clds[C_PB1 + nt * 16 + ncol];
#pragma unroll
    for (int r = 0; r < 8; ++r) {
      float v = fmaxf(acc[r] + pb, 0.0f);
      stg[SH1 + (mrow + r) * 64 + nt * 16 + ncol] = (_Float16)v;
    }
  }
  v16h a0 = load_a(stg + SH1, 64, 0, lane);
  v16h a1 = load_a(stg + SH1, 64, 1, lane);
  float* x2f = (float*)(stg + SC1);        // 512 f32
  float* pred = x2f + 512;                 // 16 f32
#pragma unroll
  for (int nt = 0; nt < 2; ++nt) {
    v8f acc = wmma32(a0, loadB(wlds + OFF_P2, 2, 0, nt, lane), zero8());
    acc = wmma32(a1, loadB(wlds + OFF_P2, 2, 1, nt, lane), acc);
    float pb = clds[C_PB2 + nt * 16 + ncol];
#pragma unroll
    for (int r = 0; r < 8; ++r)
      x2f[(mrow + r) * 32 + nt * 16 + ncol] = fmaxf(acc[r] + pb, 0.0f);
  }
  if (lane < 16) {
    float s = clds[C_BO];
#pragma unroll
    for (int k = 0; k < 32; ++k) s += x2f[lane * 32 + k] * clds[C_WO + k];
    pred[lane] = s;
    out[(size_t)(rowBase + lane) * kT + t] = s;
  }
}

__global__ __launch_bounds__(NTHR) void arlstm_kernel(
    const float* __restrict__ x,
    const float* __restrict__ W1, const float* __restrict__ Uh1, const float* __restrict__ b1,
    const float* __restrict__ W2, const float* __restrict__ Uh2, const float* __restrict__ b2,
    const float* __restrict__ Wd, const float* __restrict__ Uhd, const float* __restrict__ bd,
    const float* __restrict__ P1, const float* __restrict__ pb1,
    const float* __restrict__ P2, const float* __restrict__ pb2,
    const float* __restrict__ Wo, const float* __restrict__ bo,
    float* __restrict__ out) {
  __shared__ alignas(32) _Float16 wlds[W_TOT];
  __shared__ float clds[C_TOT];
  __shared__ alignas(32) _Float16 stg_all[WAVES][STG];

  const int tid  = threadIdx.x;
  const int lane = tid & 31;
  const int wave = tid >> 5;

  // one-time cooperative weight staging (swizzled to WMMA B-frag layout, f16)
  fill_blocks(wlds + OFF_W1,  W1,  96, 256, tid, NTHR);
  fill_blocks(wlds + OFF_UH1, Uh1, 64, 256, tid, NTHR);
  fill_blocks(wlds + OFF_W2,  W2,  64, 128, tid, NTHR);
  fill_blocks(wlds + OFF_UH2, Uh2, 32, 128, tid, NTHR);
  fill_blocks(wlds + OFF_UHD, Uhd, 32, 128, tid, NTHR);
  fill_blocks(wlds + OFF_P1,  P1,  32,  64, tid, NTHR);
  fill_blocks(wlds + OFF_P2,  P2,  64,  32, tid, NTHR);
  copyc(clds + C_B1,  b1, 256, tid, NTHR);
  copyc(clds + C_B2,  b2, 128, tid, NTHR);
  copyc(clds + C_BD,  bd, 128, tid, NTHR);
  copyc(clds + C_PB1, pb1, 64, tid, NTHR);
  copyc(clds + C_PB2, pb2, 32, tid, NTHR);
  copyc(clds + C_WD,  Wd, 128, tid, NTHR);
  copyc(clds + C_WO,  Wo,  32, tid, NTHR);
  copyc(clds + C_BO,  bo,   1, tid, NTHR);
  __syncthreads();

  _Float16* stg = stg_all[wave];
  float* c1f = (float*)(stg + SC1);   // 4 chunks x 32 lanes x 8 f32 (lane-packed)

  // zero whole per-wave stage (wave-private; same-wave LDS ops stay ordered)
  {
    v8h zh;
#pragma unroll
    for (int i = 0; i < 8; ++i) zh[i] = (_Float16)0.0f;
    for (int i = lane * 8; i < STG; i += 32 * 8) *(v8h*)(stg + i) = zh;
  }

  const int rowBase = (blockIdx.x * WAVES + wave) * ROWS;
  const int ncol = lane & 15;
  const int mrow = (lane >> 4) << 3;  // C/D layout: M = mrow + r, N = ncol

  v8f c2[2];
#pragma unroll
  for (int i = 0; i < 2; ++i) c2[i] = zero8();

  // each lane reads its own batch row (lane & 15) directly from global
  const float* xrow = x + (size_t)(rowBase + (lane & 15)) * kS * kF;

  // ---------------- encoder: 168 fused steps ----------------
  for (int t = 0; t < kS; ++t) {
    const float* xp = xrow + (size_t)t * kF;
    if (t + 1 < kS) __builtin_prefetch(xp + kF + ((lane & 16) ? 8 : 0), 0, 1);

    // A fragments for this step (x_t straight from HBM, previous h1 from LDS)
    v16h ax0 = load_xfrag(xp, 0, lane);
    v16h ax1 = load_xfrag(xp, 1, lane);
    v16h ax2 = load_xfrag(xp, 2, lane);
    v16h ah0 = load_a(stg + SH1, 64, 0, lane);
    v16h ah1 = load_a(stg + SH1, 64, 1, lane);

    // ---- layer 1: 64 units in 4 chunks of 16 (NOT unrolled: caps VGPR use) ----
#pragma unroll 1
    for (int ch = 0; ch < 4; ++ch) {
      v8f acc[4];
#pragma unroll
      for (int g = 0; g < 4; ++g) {
        const int nt = g * 4 + ch;
        v8f a = wmma32(ax0, loadB(wlds + OFF_W1, 3, 0, nt, lane), zero8());
        a = wmma32(ax1, loadB(wlds + OFF_W1, 3, 1, nt, lane), a);
        a = wmma32(ax2, loadB(wlds + OFF_W1, 3, 2, nt, lane), a);
        a = wmma32(ah0, loadB(wlds + OFF_UH1, 2, 0, nt, lane), a);
        a = wmma32(ah1, loadB(wlds + OFF_UH1, 2, 1, nt, lane), a);
        acc[g] = a;
      }
      const float bi = clds[C_B1 + 0 * 64 + ch * 16 + ncol];
      const float bf = clds[C_B1 + 1 * 64 + ch * 16 + ncol];
      const float bg = clds[C_B1 + 2 * 64 + ch * 16 + ncol];
      const float bq = clds[C_B1 + 3 * 64 + ch * 16 + ncol];
      v8f cc = *(const v8f*)(c1f + ch * 256 + lane * 8);   // cell state from LDS
#pragma unroll
      for (int r = 0; r < 8; ++r) {
        float c = sigf(acc[1][r] + bf) * cc[r] +
                  sigf(acc[0][r] + bi) * tanhc(acc[2][r] + bg);
        cc[r] = c;
        float h = sigf(acc[3][r] + bq) * tanhc(c);
        stg[SH1 + (mrow + r) * 64 + ch * 16 + ncol] = (_Float16)h;
      }
      *(v8f*)(c1f + ch * 256 + lane * 8) = cc;
    }

    // ---- layer 2: 32 units in 2 chunks of 16 ----
    v16h an0 = load_a(stg + SH1, 64, 0, lane);       // new h1
    v16h an1 = load_a(stg + SH1, 64, 1, lane);
    v16h ah2 = load_a(stg + SH2, 32, 0, lane);       // previous h2
#pragma unroll
    for (int ch = 0; ch < 2; ++ch) {
      v8f acc[4];
#pragma unroll
      for (int g = 0; g < 4; ++g) {
        const int nt = g * 2 + ch;
        v8f a = wmma32(an0, loadB(wlds + OFF_W2, 2, 0, nt, lane), zero8());
        a = wmma32(an1, loadB(wlds + OFF_W2, 2, 1, nt, lane), a);
        a = wmma32(ah2, loadB(wlds + OFF_UH2, 1, 0, nt, lane), a);
        acc[g] = a;
      }
      const float bi = clds[C_B2 + 0 * 32 + ch * 16 + ncol];
      const float bf = clds[C_B2 + 1 * 32 + ch * 16 + ncol];
      const float bg = clds[C_B2 + 2 * 32 + ch * 16 + ncol];
      const float bq = clds[C_B2 + 3 * 32 + ch * 16 + ncol];
#pragma unroll
      for (int r = 0; r < 8; ++r) {
        float c = sigf(acc[1][r] + bf) * c2[ch][r] +
                  sigf(acc[0][r] + bi) * tanhc(acc[2][r] + bg);
        c2[ch][r] = c;
        float h = sigf(acc[3][r] + bq) * tanhc(c);
        stg[SH2 + (mrow + r) * 32 + ch * 16 + ncol] = (_Float16)h;
      }
    }
  }

  // ---------------- warmup prediction (t = 0) ----------------
  project_store(stg, wlds, clds, lane, out, rowBase, 0);

  // ---------------- autoregressive decode (t = 1..23) ----------------
#pragma unroll 1
  for (int t = 1; t < kT; ++t) {
    v16h ah2 = load_a(stg + SH2, 32, 0, lane);
    const float* pred = (const float*)(stg + SC1) + 512;
    float pm[8];
#pragma unroll
    for (int r = 0; r < 8; ++r) pm[r] = pred[mrow + r];

#pragma unroll
    for (int ch = 0; ch < 2; ++ch) {
      v8f acc[4];
#pragma unroll
      for (int g = 0; g < 4; ++g)
        acc[g] = wmma32(ah2, loadB(wlds + OFF_UHD, 1, 0, g * 2 + ch, lane), zero8());
      float z[4][8];
#pragma unroll
      for (int g = 0; g < 4; ++g) {
        const int col = g * 32 + ch * 16 + ncol;
        const float wd = clds[C_WD + col];
        const float bb = clds[C_BD + col];
#pragma unroll
        for (int r = 0; r < 8; ++r) z[g][r] = acc[g][r] + pm[r] * wd + bb;
      }
#pragma unroll
      for (int r = 0; r < 8; ++r) {
        float c = sigf(z[1][r]) * c2[ch][r] + sigf(z[0][r]) * tanhc(z[2][r]);
        c2[ch][r] = c;
        float h = sigf(z[3][r]) * tanhc(c);
        stg[SH2 + (mrow + r) * 32 + ch * 16 + ncol] = (_Float16)h;
      }
    }
    project_store(stg, wlds, clds, lane, out, rowBase, t);
  }
}

extern "C" void kernel_launch(void* const* d_in, const int* in_sizes, int n_in,
                              void* d_out, int out_size, void* d_ws, size_t ws_size,
                              hipStream_t stream) {
  (void)in_sizes; (void)n_in; (void)out_size; (void)d_ws; (void)ws_size;
  const float* x   = (const float*)d_in[0];
  const float* W1  = (const float*)d_in[1];
  const float* Uh1 = (const float*)d_in[2];
  const float* b1  = (const float*)d_in[3];
  const float* W2  = (const float*)d_in[4];
  const float* Uh2 = (const float*)d_in[5];
  const float* b2  = (const float*)d_in[6];
  const float* Wd  = (const float*)d_in[7];
  const float* Uhd = (const float*)d_in[8];
  const float* bd  = (const float*)d_in[9];
  const float* P1  = (const float*)d_in[10];
  const float* pb1 = (const float*)d_in[11];
  const float* P2  = (const float*)d_in[12];
  const float* pb2 = (const float*)d_in[13];
  const float* Wo  = (const float*)d_in[14];
  const float* bo  = (const float*)d_in[15];
  float* out = (float*)d_out;

  dim3 grid(kB / (WAVES * ROWS));  // 128 workgroups
  dim3 block(NTHR);                // 128 threads = 4 wave32
  arlstm_kernel<<<grid, block, 0, stream>>>(x, W1, Uh1, b1, W2, Uh2, b2, Wd, Uhd,
                                            bd, P1, pb1, P2, pb2, Wo, bo, out);
}
